// SelfAttention1DLayer_50285477101625
// MI455X (gfx1250) — matrix operand order repaired and checked by reading, not compiled, not run on hardware
//
#include <hip/hip_runtime.h>
#include <hip/hip_bf16.h>

// Problem constants (from reference)
constexpr int Bc  = 16;
constexpr int Tc  = 2048;
constexpr int Dc  = 512;
constexpr int Rc  = 2048;
constexpr int DAc = 512;
constexpr int CHUNK = 64;            // flash chunk along T
constexpr int NCH   = Tc / CHUNK;    // 32 chunks

typedef __attribute__((ext_vector_type(16))) __bf16       v16bf;
typedef __attribute__((ext_vector_type(8)))  float        v8f;
typedef __attribute__((ext_vector_type(4)))  unsigned int u32x4;
typedef __attribute__((ext_vector_type(8)))  int          i32x8;
typedef __attribute__((ext_vector_type(4)))  int          i32x4;

#if defined(__has_builtin)
#if __has_builtin(__builtin_amdgcn_tensor_load_to_lds)
#define HAVE_TDM 1
#endif
#endif

// ---------------------------------------------------------------------------
// Fast tanh: gfx1250 has a hardware v_tanh_f32 (TRANS op); use it if exposed.
// ---------------------------------------------------------------------------
__device__ __forceinline__ float fast_tanh(float v) {
#if defined(__has_builtin)
#if __has_builtin(__builtin_amdgcn_tanhf)
  return __builtin_amdgcn_tanhf(v);
#elif __has_builtin(__builtin_amdgcn_tanh_f32)
  return __builtin_amdgcn_tanh_f32(v);
#else
  return tanhf(v);
#endif
#else
  return tanhf(v);
#endif
}

// ---------------------------------------------------------------------------
// WMMA helper: D = A(16x32 bf16) x B(32x16 bf16) + C(16x16 f32), wave32
// ---------------------------------------------------------------------------
__device__ __forceinline__ v8f wmma_bf16(v16bf a, v16bf b, v8f c) {
  return __builtin_amdgcn_wmma_f32_16x16x32_bf16(
      /*neg_a=*/false, a, /*neg_b=*/false, b,
      /*c_mod=*/(short)0, c, /*reuse_a=*/false, /*reuse_b=*/false);
}

// A-fragment (16x32, bf16 source, row-major, stride in elements).
// Lane l: row m=l&15, half h=l>>4. elems 0..7 -> K=k0+h*8+j ; 8..15 -> K=k0+16+h*8+j
__device__ __forceinline__ v16bf load_a_b(const __bf16* base, int stride, int k0, int lane) {
  int m = lane & 15, h = lane >> 4;
  const __bf16* r = base + (size_t)m * stride + k0 + h * 8;
  v16bf a;
  *((uint4*)&a)       = *(const uint4*)(r);
  *(((uint4*)&a) + 1) = *(const uint4*)(r + 16);
  return a;
}

// A-fragment from fp32 source with convert (row-major fp32, stride in elements)
__device__ __forceinline__ v16bf load_a_f(const float* base, int stride, int k0, int lane) {
  int m = lane & 15, h = lane >> 4;
  const float* r = base + (size_t)m * stride;
  v16bf a;
#pragma unroll
  for (int j = 0; j < 8; ++j) a[j] = (__bf16)r[k0 + h * 8 + j];
#pragma unroll
  for (int j = 0; j < 8; ++j) a[8 + j] = (__bf16)r[k0 + 16 + h * 8 + j];
  return a;
}

// B-fragment where memory holds B^T row-major (B[k][n] = mem[n][k]).
// Lane l: col n=l&15, h=l>>4. elems 0..7 -> K=k0+h*8+j ; 8..15 -> K=k0+16+h*8+j
__device__ __forceinline__ v16bf load_bT_b(const __bf16* base, int stride, int k0, int lane) {
  int n = lane & 15, h = lane >> 4;
  const __bf16* r = base + (size_t)n * stride + k0 + h * 8;
  v16bf b;
  *((uint4*)&b)       = *(const uint4*)(r);
  *(((uint4*)&b) + 1) = *(const uint4*)(r + 16);
  return b;
}

// ---------------------------------------------------------------------------
// LDS byte-offset of a generic pointer (generic -> addrspace(3) -> offset)
// ---------------------------------------------------------------------------
typedef __attribute__((address_space(3))) const void* lds_cptr_t;
__device__ __forceinline__ unsigned lds_off_of(const void* p) {
  return (unsigned)(size_t)(lds_cptr_t)p;
}

// ---------------------------------------------------------------------------
// TDM: 1-D contiguous copy of `elems` 2-byte elements from global to LDS.
// D# per CDNA5 ISA ch.8: group0 {count=1, lds_addr, global_addr, type=2},
// group1 {data_size=2B, tensor_dim0=elems, tensor_dim1=1, tile_dim0=elems,
//         tile_dim1=1, dim0_stride=elems}.
// ---------------------------------------------------------------------------
#ifdef HAVE_TDM
__device__ __forceinline__ void tdm_load_linear(unsigned lds_off, const void* gptr,
                                                unsigned elems) {
  unsigned long long ga = (unsigned long long)gptr;
  u32x4 g0;
  g0[0] = 1u;                                                   // count=1, no gather
  g0[1] = lds_off;                                              // lds_addr
  g0[2] = (unsigned)ga;                                         // global_addr[31:0]
  g0[3] = ((unsigned)(ga >> 32) & 0x01FFFFFFu) | (2u << 30);    // addr[56:32] | type=2
  i32x8 g1;
  g1[0] = (int)(1u << 16);                     // data_size = 2 bytes
  g1[1] = (int)((elems & 0xFFFFu) << 16);      // tensor_dim0[15:0]
  g1[2] = (int)((elems >> 16) | (1u << 16));   // tensor_dim0[31:16] | tensor_dim1=1
  g1[3] = (int)((elems & 0xFFFFu) << 16);      // tensor_dim1 hi(0)  | tile_dim0
  g1[4] = 1;                                   // tile_dim1 = 1, tile_dim2 = 0
  g1[5] = (int)elems;                          // tensor_dim0_stride[31:0]
  g1[6] = 0;                                   // stride0 hi | stride1 lo
  g1[7] = 0;
  i32x4 z4 = {0, 0, 0, 0};
#if __clang_major__ >= 23
  i32x8 z8 = {0, 0, 0, 0, 0, 0, 0, 0};
  __builtin_amdgcn_tensor_load_to_lds(g0, g1, z4, z4, z8, 0);
#else
  __builtin_amdgcn_tensor_load_to_lds(g0, g1, z4, z4, 0);
#endif
}
#endif

// Stage one Y chunk (CHUNK rows x DAc bf16, contiguous 64KB) into LDS.
__device__ __forceinline__ void stage_y_chunk(const __bf16* src, __bf16* dstLDS,
                                              int wave, int tid) {
#ifdef HAVE_TDM
  if (wave == 0) tdm_load_linear(lds_off_of(dstLDS), src, (unsigned)(CHUNK * DAc));
#else
  const uint4* s = (const uint4*)src;
  uint4*       d = (uint4*)dstLDS;
  for (int i = tid; i < CHUNK * DAc / 8; i += 128) d[i] = s[i];
#endif
}

__device__ __forceinline__ void wait_y_chunk(int wave) {
#ifdef HAVE_TDM
  if (wave == 0) __builtin_amdgcn_s_wait_tensorcnt(0);
#endif
}

// ---------------------------------------------------------------------------
// Kernel 1: xT[b][d][t] = bf16(x[b][t][d])   (B,D,T) row-major
// ---------------------------------------------------------------------------
__global__ void transpose_x_kernel(const float* __restrict__ x, __bf16* __restrict__ xT) {
  __shared__ __bf16 tile[32][33];
  const int b  = blockIdx.z;
  const int t0 = blockIdx.x * 32;
  const int d0 = blockIdx.y * 32;
  const int tx = threadIdx.x;
  const int ty = threadIdx.y;
#pragma unroll
  for (int i = 0; i < 4; ++i) {
    int t = t0 + ty + i * 8;
    tile[ty + i * 8][tx] = (__bf16)x[((size_t)b * Tc + t) * Dc + d0 + tx];
  }
  __syncthreads();
#pragma unroll
  for (int i = 0; i < 4; ++i) {
    int d = d0 + ty + i * 8;
    xT[((size_t)b * Dc + d) * Tc + t0 + tx] = tile[tx][ty + i * 8];
  }
}

// ---------------------------------------------------------------------------
// Kernel 2: Y[b][t][a] = bf16( tanh( sum_d wk[a][d] * x[b][t][d] ) )
// C(64 T-rows x 64 DA-cols) per workgroup; wk tile staged in LDS as bf16.
// ---------------------------------------------------------------------------
__global__ void compute_y_kernel(const float* __restrict__ x,
                                 const float* __restrict__ wk,
                                 __bf16* __restrict__ y) {
  __shared__ alignas(16) __bf16 wk_s[64 * Dc];   // 64 KB
  const int lane = threadIdx.x & 31;
  const int wave = threadIdx.x >> 5;
  const int b  = blockIdx.z;
  const int t0 = blockIdx.y * 64 + wave * 16;
  const int a0 = blockIdx.x * 64;

  for (int i = threadIdx.x; i < 64 * Dc; i += 128)
    wk_s[i] = (__bf16)wk[(size_t)(a0 + (i >> 9)) * Dc + (i & (Dc - 1))];
  __syncthreads();

  const float* xrow = x + ((size_t)b * Tc + t0) * Dc;

  v8f acc[4] = {v8f{}, v8f{}, v8f{}, v8f{}};
#pragma unroll 4
  for (int k0 = 0; k0 < Dc; k0 += 32) {
    v16bf af = load_a_f(xrow, Dc, k0, lane);
#pragma unroll
    for (int nt = 0; nt < 4; ++nt) {
      v16bf bf = load_bT_b(wk_s + (size_t)(nt * 16) * Dc, Dc, k0, lane);
      acc[nt] = wmma_bf16(af, bf, acc[nt]);
    }
  }
  const int col = lane & 15, h = lane >> 4;
#pragma unroll
  for (int nt = 0; nt < 4; ++nt)
#pragma unroll
    for (int v = 0; v < 8; ++v) {
      int tt = t0 + h * 8 + v;
      int aa = a0 + nt * 16 + col;
      y[((size_t)b * Tc + tt) * DAc + aa] = (__bf16)fast_tanh(acc[nt][v]);
    }
}

// ---------------------------------------------------------------------------
// Kernel 3: fused  sim = kern_tile @ Y^T -> online softmax -> out += P @ x
// One WG = (batch, 16 R-rows); 4 waves; wave w owns D cols [w*128, w*128+128).
// Y chunks TDM-staged into double-buffered LDS.
// ---------------------------------------------------------------------------
__global__ void attn_kernel(const float*  __restrict__ kern,   // (R, DA) fp32
                            const __bf16* __restrict__ Y,      // (B, T, DA) bf16
                            const __bf16* __restrict__ xT,     // (B, D, T) bf16
                            float* __restrict__ out) {         // (B, R, D) fp32
  __shared__ alignas(16) __bf16 ybuf[2 * CHUNK * DAc];  // 128 KB double buffer
  __shared__ alignas(16) __bf16 kern_s[16 * DAc];       // 16 KB
  __shared__ alignas(16) float  s_tile[16 * CHUNK];     // 4 KB
  __shared__ alignas(16) __bf16 p_tile[16 * CHUNK];     // 2 KB
  __shared__ float red[16][8];
  __shared__ float row_m[16], row_l[16], row_scale[16];

  const int tid  = threadIdx.x;
  const int lane = tid & 31;
  const int wave = tid >> 5;
  const int b  = blockIdx.y;
  const int r0 = blockIdx.x * 16;
  const int h  = lane >> 4;
  const int nc = lane & 15;

  // Stage the 16x512 kernel tile (fp32 -> bf16) into LDS once.
  for (int i = tid; i < 16 * DAc; i += 128)
    kern_s[i] = (__bf16)kern[(size_t)(r0 + (i >> 9)) * DAc + (i & (DAc - 1))];
  if (tid < 16) { row_m[tid] = -3.0e38f; row_l[tid] = 0.0f; }

  v8f acc[8];
#pragma unroll
  for (int nt = 0; nt < 8; ++nt) acc[nt] = v8f{};

  const __bf16* Yb = Y + (size_t)b * Tc * DAc;

  // Prologue: kick off chunk 0 -> buffer 0
  stage_y_chunk(Yb, ybuf, wave, tid);

  for (int tcb = 0; tcb < NCH; ++tcb) {
    const int buf = tcb & 1;
    wait_y_chunk(wave);           // chunk tcb landed in ybuf[buf]
    __syncthreads();
    if (tcb + 1 < NCH)            // prefetch chunk tcb+1 into the other buffer
      stage_y_chunk(Yb + (size_t)(tcb + 1) * CHUNK * DAc,
                    ybuf + (size_t)(buf ^ 1) * CHUNK * DAc, wave, tid);

    // ---- Phase 1: each wave computes a 16x16 sub-tile of S (K = DA = 512)
    const __bf16* yrows = ybuf + (size_t)buf * CHUNK * DAc + (size_t)(wave * 16) * DAc;
    v8f sacc = v8f{};
#pragma unroll 4
    for (int k0 = 0; k0 < DAc; k0 += 32) {
      v16bf af = load_a_b(kern_s, DAc, k0, lane);
      v16bf bf = load_bT_b(yrows, DAc, k0, lane);
      sacc = wmma_bf16(af, bf, sacc);
    }
    {
      const int col = wave * 16 + nc;
#pragma unroll
      for (int v = 0; v < 8; ++v)
        s_tile[(h * 8 + v) * CHUNK + col] = sacc[v];
    }
    __syncthreads();

    // ---- Phase 2: online softmax update (8 threads per row)
    {
      const int r  = tid & 15;
      const int cs = (tid >> 4) * 8;
      float pm = -3.0e38f;
#pragma unroll
      for (int j = 0; j < 8; ++j) pm = fmaxf(pm, s_tile[r * CHUNK + cs + j]);
      red[r][tid >> 4] = pm;
      __syncthreads();
      if (tid < 16) {
        float mold = row_m[tid];
        float mc = mold;
#pragma unroll
        for (int k = 0; k < 8; ++k) mc = fmaxf(mc, red[tid][k]);
        row_scale[tid] = __expf(mold - mc);
        row_m[tid] = mc;
      }
      __syncthreads();
      const float mc = row_m[r];
      float ps = 0.0f;
#pragma unroll
      for (int j = 0; j < 8; ++j) {
        float p = __expf(s_tile[r * CHUNK + cs + j] - mc);
        p_tile[r * CHUNK + cs + j] = (__bf16)p;
        ps += p;
      }
      red[r][tid >> 4] = ps;
      __syncthreads();
      if (tid < 16) {
        float l = row_l[tid] * row_scale[tid];
#pragma unroll
        for (int k = 0; k < 8; ++k) l += red[tid][k];
        row_l[tid] = l;
      }
      __syncthreads();
    }

    // ---- Phase 3: rescale accumulators, accumulate P @ x  (K = CHUNK = 64)
    if (tcb + 1 < NCH)
      __builtin_prefetch(xT + ((size_t)b * Dc + wave * 128) * Tc + (size_t)(tcb + 1) * CHUNK,
                         0, 1);
    float sc[8];
#pragma unroll
    for (int v = 0; v < 8; ++v) sc[v] = row_scale[h * 8 + v];
#pragma unroll
    for (int nt = 0; nt < 8; ++nt)
#pragma unroll
      for (int v = 0; v < 8; ++v) acc[nt][v] *= sc[v];

#pragma unroll
    for (int k0 = 0; k0 < CHUNK; k0 += 32) {
      v16bf af = load_a_b(p_tile, CHUNK, k0, lane);
#pragma unroll
      for (int nt = 0; nt < 8; ++nt) {
        const __bf16* xrow =
            xT + ((size_t)b * Dc + wave * 128 + nt * 16) * Tc + (size_t)tcb * CHUNK;
        v16bf bf = load_bT_b(xrow, Tc, k0, lane);
        acc[nt] = wmma_bf16(af, bf, acc[nt]);
      }
    }
    __syncthreads();
  }

  // ---- Finalize: divide by softmax denominator, store fp32
  float linv[8];
#pragma unroll
  for (int v = 0; v < 8; ++v) linv[v] = 1.0f / row_l[h * 8 + v];
#pragma unroll
  for (int nt = 0; nt < 8; ++nt) {
    const int dcol = wave * 128 + nt * 16 + nc;
#pragma unroll
    for (int v = 0; v < 8; ++v)
      out[((size_t)b * Rc + r0 + h * 8 + v) * Dc + dcol] = acc[nt][v] * linv[v];
  }
}

// ---------------------------------------------------------------------------
extern "C" void kernel_launch(void* const* d_in, const int* in_sizes, int n_in,
                              void* d_out, int out_size, void* d_ws, size_t ws_size,
                              hipStream_t stream) {
  const float* x    = (const float*)d_in[0];   // (B, T, D)
  const float* kern = (const float*)d_in[1];   // (R, DA)
  const float* wk   = (const float*)d_in[2];   // (DA, D)
  float* out = (float*)d_out;                  // (B, R, D)

  __bf16* Yw  = (__bf16*)d_ws;                  // (B, T, DA) bf16 : 32 MB
  __bf16* xTw = Yw + (size_t)Bc * Tc * DAc;     // (B, D, T)  bf16 : 32 MB

  transpose_x_kernel<<<dim3(Tc / 32, Dc / 32, Bc), dim3(32, 8), 0, stream>>>(x, xTw);
  compute_y_kernel<<<dim3(DAc / 64, Tc / 64, Bc), dim3(128), 0, stream>>>(x, wk, Yw);
  attn_kernel<<<dim3(Rc / 16, Bc), dim3(128), 0, stream>>>(kern, Yw, xTw, out);
}
